// ContrastiveLoss_65764539236396
// MI455X (gfx1250) — compile-verified, matrix operand order checked by reading, not run on hardware
//
#include <hip/hip_runtime.h>

// ---------------------------------------------------------------------------
// Types for CDNA5 WMMA (gfx1250, wave32)
// ---------------------------------------------------------------------------
typedef __attribute__((ext_vector_type(16))) __bf16 v16bf;
typedef __attribute__((ext_vector_type(8)))  __bf16 v8bf;
typedef __attribute__((ext_vector_type(8)))  float  v8f;

#define N_IMG 64
#define N_CAP 64
#define FRAMES 64
#define WORDS 40
#define WPAD 48
#define DIM 512
#define LAMBDA_SOFTMAX 9.0f
#define LAMBDA_LSE 6.0f
#define MARGIN 0.2f
#define EPS 1e-8f

// Workspace byte offsets.  Captions are staged ZERO-PADDED to 48 rows so the
// WMMA fragment loads need no bounds checks at all.
//   imgs_bf : 64*64*512*2      = 4,194,304   @ 0
//   caps_bf : 64*48*512*2      = 3,145,728*2 @ 4,194,304
//   n1      : 64*64*4          = 16,384      @ 10,485,760
//   G       : 64*48*48*4       = 589,824     @ 10,502,144
//   S       : 64*64*4          = 16,384      @ 11,091,968   (total ~11.1 MB)
#define OFF_CAPS  4194304
#define OFF_N1    10485760
#define OFF_G     10502144
#define OFF_S     11091968

// ---------------------------------------------------------------------------
// Unconditional bf16 fragment load (16-bit A/B operand per ISA layout).
// Lane l holds row (l&15) of the tile; its 16 elements are two contiguous
// 8-element chunks at k offsets (l>>4)*8 and (l>>4)*8 + 16.  The caller
// pre-bakes the per-lane row/half offset into the pointer.
// ---------------------------------------------------------------------------
__device__ __forceinline__ v16bf ldfrag(const __bf16* p) {
    v8bf lo = *(const v8bf*)(p);
    v8bf hi = *(const v8bf*)(p + 16);
    return __builtin_shufflevector(lo, hi, 0, 1, 2, 3, 4, 5, 6, 7,
                                           8, 9, 10, 11, 12, 13, 14, 15);
}

__device__ __forceinline__ float leaky01(float x) {
    return x > 0.0f ? x : 0.1f * x;
}

// ---------------------------------------------------------------------------
// Kernel 1: f32 -> bf16 staging.  Captions padded to 48 zero rows per sample.
// ---------------------------------------------------------------------------
__global__ void convert_kernel(const float* __restrict__ images,
                               const float* __restrict__ captions,
                               __bf16* __restrict__ imgs_bf,
                               __bf16* __restrict__ caps_bf,
                               int nimg, int ncap_pad) {
    int idx = blockIdx.x * 256 + threadIdx.x;
    if (idx < nimg) imgs_bf[idx] = (__bf16)images[idx];
    if (idx < ncap_pad) {
        int j = idx / (WPAD * DIM);
        int rem = idx - j * (WPAD * DIM);
        int w = rem >> 9;               // rem / DIM
        int d = rem & (DIM - 1);
        float v = (w < WORDS) ? captions[((size_t)j * WORDS + w) * DIM + d]
                              : 0.0f;
        caps_bf[idx] = (__bf16)v;
    }
}

// ---------------------------------------------------------------------------
// Kernel 2: per-frame image norms n1[i,f] = ||images[i,f,:]||.
// ---------------------------------------------------------------------------
__global__ void norms_kernel(const float* __restrict__ images,
                             float* __restrict__ n1) {
    __shared__ float red[64];
    int row = blockIdx.x;                       // row = i*FRAMES + f
    const float* p = images + (size_t)row * DIM;
    float s = 0.0f;
    for (int k = threadIdx.x; k < DIM; k += 64) {
        float v = p[k];
        s += v * v;
    }
    red[threadIdx.x] = s;
    __syncthreads();
    if (threadIdx.x == 0) {
        float t = 0.0f;
        for (int k = 0; k < 64; ++k) t += red[k];
        n1[row] = sqrtf(t);
    }
}

// ---------------------------------------------------------------------------
// Kernel 3: per-caption Gram G_j = cap_j * cap_j^T (48x48; pad rows are zero
// in the staged buffer, so products involving them are naturally zero).
// Waves 0..2 each own one 16-wide column tile and 3 row-tile accumulators.
// ---------------------------------------------------------------------------
__global__ void gram_kernel(const __bf16* __restrict__ caps_bf,
                            float* __restrict__ G) {
    int j = blockIdx.x;
    int lane = threadIdx.x & 31;
    int wv = threadIdx.x >> 5;
    const __bf16* capj = caps_bf + (size_t)j * WPAD * DIM;
    float* Gj = G + (size_t)j * WPAD * WPAD;

    if (wv < 3) {                                // wave-uniform branch
        int nt = wv;
        int laneRow = lane & 15;
        int kHalf = (lane >> 4) << 3;
        const __bf16* bptr = capj + (size_t)(nt * 16 + laneRow) * DIM + kHalf;
        const __bf16* aptr0 = capj + (size_t)(0 + laneRow) * DIM + kHalf;
        const __bf16* aptr1 = capj + (size_t)(16 + laneRow) * DIM + kHalf;
        const __bf16* aptr2 = capj + (size_t)(32 + laneRow) * DIM + kHalf;

        v8f acc[3];
        #pragma unroll
        for (int m = 0; m < 3; ++m)
            #pragma unroll
            for (int q = 0; q < 8; ++q) acc[m][q] = 0.0f;

        for (int k = 0; k < DIM; k += 32) {
            v16bf b = ldfrag(bptr + k);
            v16bf a0 = ldfrag(aptr0 + k);
            v16bf a1 = ldfrag(aptr1 + k);
            v16bf a2 = ldfrag(aptr2 + k);
            acc[0] = __builtin_amdgcn_wmma_f32_16x16x32_bf16(
                false, a0, false, b, (short)0, acc[0], false, false);
            acc[1] = __builtin_amdgcn_wmma_f32_16x16x32_bf16(
                false, a1, false, b, (short)0, acc[1], false, false);
            acc[2] = __builtin_amdgcn_wmma_f32_16x16x32_bf16(
                false, a2, false, b, (short)0, acc[2], false, false);
        }
        int n = nt * 16 + laneRow;
        int half = lane >> 4;
        #pragma unroll
        for (int m = 0; m < 3; ++m)
            #pragma unroll
            for (int r = 0; r < 8; ++r)
                Gj[(m * 16 + r + 8 * half) * WPAD + n] = acc[m][r];
    }
}

// ---------------------------------------------------------------------------
// Kernel 4: one workgroup (4 wave32) per (caption j, image i) pair.
//   Phase 1 : Araw[w,f] = cap_j[w,:] . img_i[f,:]   (WMMA bf16 48x64x512;
//             each wave owns one frame-column tile, 3 accumulators)
//   Phase 2a: leaky-ReLU + l2 normalize over frames (per word)
//   Phase 2b: softmax over words (temp 9), then
//             num[f]  = sum_w aw[f,w] * Araw[w,f]
//             n2sq[f] = aw[f,:]^T G_j aw[f,:]        (Gram trick; no wctx)
//             sim, unstabilized LSE over valid frames -> S[j,i]
// ---------------------------------------------------------------------------
__global__ void pair_kernel(const __bf16* __restrict__ imgs_bf,
                            const __bf16* __restrict__ caps_bf,
                            const float* __restrict__ n1,
                            const float* __restrict__ G,
                            const int* __restrict__ img_lens,
                            const int* __restrict__ cap_lens,
                            float* __restrict__ S) {
    __shared__ float Araw[WPAD * FRAMES];   // [w][f]  12 KB
    __shared__ float AW[FRAMES * WPAD];     // [f][w]  12 KB
    __shared__ float Gs[WPAD * WPAD];       //          9 KB
    __shared__ float red[FRAMES];

    int pid = blockIdx.x;
    int j = pid >> 6;          // caption index (row of S)
    int i = pid & 63;          // image index  (col of S)
    int lane = threadIdx.x & 31;
    int wv = threadIdx.x >> 5;
    int capLen = cap_lens[j];
    int imgLen = img_lens[i];

    const __bf16* capj = caps_bf + (size_t)j * WPAD * DIM;
    const __bf16* imgi = imgs_bf + (size_t)i * FRAMES * DIM;

    // Warm L0/L1 for this pair's tiles (all data is L2-resident).
    for (int q = threadIdx.x * 128; q < WPAD * DIM * 2; q += 128 * 128)
        __builtin_prefetch((const char*)capj + q, 0, 0);
    for (int q = threadIdx.x * 128; q < FRAMES * DIM * 2; q += 128 * 128)
        __builtin_prefetch((const char*)imgi + q, 0, 0);

    // Stage Gram matrix for caption j into LDS.
    for (int q = threadIdx.x; q < WPAD * WPAD; q += 128)
        Gs[q] = G[(size_t)j * WPAD * WPAD + q];

    // ---- Phase 1: Araw via WMMA.  Wave wv owns frame columns [16wv,16wv+16).
    {
        int laneRow = lane & 15;
        int half = lane >> 4;
        int kHalf = half << 3;
        const __bf16* bptr = imgi + (size_t)(wv * 16 + laneRow) * DIM + kHalf;
        const __bf16* aptr0 = capj + (size_t)(0 + laneRow) * DIM + kHalf;
        const __bf16* aptr1 = capj + (size_t)(16 + laneRow) * DIM + kHalf;
        const __bf16* aptr2 = capj + (size_t)(32 + laneRow) * DIM + kHalf;

        v8f acc[3];
        #pragma unroll
        for (int m = 0; m < 3; ++m)
            #pragma unroll
            for (int q = 0; q < 8; ++q) acc[m][q] = 0.0f;

        for (int k = 0; k < DIM; k += 32) {
            v16bf b = ldfrag(bptr + k);
            v16bf a0 = ldfrag(aptr0 + k);
            v16bf a1 = ldfrag(aptr1 + k);
            v16bf a2 = ldfrag(aptr2 + k);
            acc[0] = __builtin_amdgcn_wmma_f32_16x16x32_bf16(
                false, a0, false, b, (short)0, acc[0], false, false);
            acc[1] = __builtin_amdgcn_wmma_f32_16x16x32_bf16(
                false, a1, false, b, (short)0, acc[1], false, false);
            acc[2] = __builtin_amdgcn_wmma_f32_16x16x32_bf16(
                false, a2, false, b, (short)0, acc[2], false, false);
        }
        int f = wv * 16 + laneRow;
        #pragma unroll
        for (int m = 0; m < 3; ++m)
            #pragma unroll
            for (int r = 0; r < 8; ++r)
                Araw[(m * 16 + r + 8 * half) * FRAMES + f] = acc[m][r];
    }
    __syncthreads();

    // ---- Phase 2a: clipped_l2norm per word row, write transposed ----
    int w = threadIdx.x;
    if (w < WPAD) {
        if (w < capLen) {
            float ss = 0.0f;
            for (int f = 0; f < imgLen; ++f) {
                float v = leaky01(Araw[w * FRAMES + f]);
                ss += v * v;
            }
            float inv = 1.0f / (sqrtf(ss) + EPS);
            for (int f = 0; f < FRAMES; ++f) {
                float v = (f < imgLen) ? leaky01(Araw[w * FRAMES + f]) * inv
                                       : 0.0f;
                AW[f * WPAD + w] = v;
            }
        } else {
            for (int f = 0; f < FRAMES; ++f) AW[f * WPAD + w] = 0.0f;
        }
    }
    __syncthreads();

    // ---- Phase 2b: softmax over words, num, Gram quadratic form, LSE ----
    int f = threadIdx.x;
    if (f < FRAMES) {
        float e = 0.0f;
        if (f < imgLen) {
            float* row = &AW[f * WPAD];
            float m = -1e30f;
            for (int ww = 0; ww < capLen; ++ww)
                m = fmaxf(m, row[ww] * LAMBDA_SOFTMAX);
            float s = 0.0f;
            for (int ww = 0; ww < capLen; ++ww) {
                float x = __expf(row[ww] * LAMBDA_SOFTMAX - m);
                row[ww] = x;
                s += x;
            }
            float invs = 1.0f / s;
            float num = 0.0f;
            for (int ww = 0; ww < capLen; ++ww) {
                float a = row[ww] * invs;
                row[ww] = a;
                num += a * Araw[ww * FRAMES + f];
            }
            float n2sq = 0.0f;
            for (int ww = 0; ww < capLen; ++ww) {
                const float* Grow = &Gs[ww * WPAD];
                float gw = 0.0f;
                for (int w2 = 0; w2 < capLen; ++w2)
                    gw += Grow[w2] * row[w2];
                n2sq += row[ww] * gw;
            }
            float denom = fmaxf(n1[(size_t)i * FRAMES + f] *
                                sqrtf(fmaxf(n2sq, 0.0f)), EPS);
            e = __expf(LAMBDA_LSE * (num / denom));
        }
        red[f] = e;
    }
    __syncthreads();
    if (threadIdx.x == 0) {
        float s = 0.0f;
        for (int k = 0; k < FRAMES; ++k) s += red[k];
        S[pid] = __logf(s) / LAMBDA_LSE;
    }
}

// ---------------------------------------------------------------------------
// Kernel 5: hinge loss with hardest negatives over the 64x64 score matrix.
// ---------------------------------------------------------------------------
__global__ void loss_kernel(const float* __restrict__ S,
                            float* __restrict__ out) {
    __shared__ float red[64];
    int t = threadIdx.x;                   // 64 threads
    float dt = S[t * 64 + t];
    float rmax = 0.0f, cmax = 0.0f;
    for (int k = 0; k < 64; ++k) {
        if (k == t) continue;
        rmax = fmaxf(rmax, fmaxf(MARGIN + S[t * 64 + k] - dt, 0.0f));
        cmax = fmaxf(cmax, fmaxf(MARGIN + S[k * 64 + t] - dt, 0.0f));
    }
    red[t] = rmax + cmax;
    __syncthreads();
    if (t == 0) {
        float s = 0.0f;
        for (int k = 0; k < 64; ++k) s += red[k];
        out[0] = s;
    }
}

// ---------------------------------------------------------------------------
// Host launcher.
// ---------------------------------------------------------------------------
extern "C" void kernel_launch(void* const* d_in, const int* in_sizes, int n_in,
                              void* d_out, int out_size, void* d_ws,
                              size_t ws_size, hipStream_t stream) {
    const float* images   = (const float*)d_in[0];
    const float* captions = (const float*)d_in[1];
    const int*   img_lens = (const int*)d_in[2];
    const int*   cap_lens = (const int*)d_in[3];
    float* out = (float*)d_out;

    char* ws = (char*)d_ws;
    __bf16* imgs_bf = (__bf16*)(ws);
    __bf16* caps_bf = (__bf16*)(ws + OFF_CAPS);
    float*  n1      = (float*)(ws + OFF_N1);
    float*  G       = (float*)(ws + OFF_G);
    float*  S       = (float*)(ws + OFF_S);

    int nimg = N_IMG * FRAMES * DIM;      // 2,097,152
    int ncap_pad = N_CAP * WPAD * DIM;    // 1,572,864

    convert_kernel<<<(nimg + 255) / 256, 256, 0, stream>>>(
        images, captions, imgs_bf, caps_bf, nimg, ncap_pad);
    norms_kernel<<<N_IMG * FRAMES, 64, 0, stream>>>(images, n1);
    gram_kernel<<<N_CAP, 128, 0, stream>>>(caps_bf, G);
    pair_kernel<<<N_CAP * N_IMG, 128, 0, stream>>>(
        imgs_bf, caps_bf, n1, G, img_lens, cap_lens, S);
    loss_kernel<<<1, 64, 0, stream>>>(S, out);
}
